// BACPI_63797444214999
// MI455X (gfx1250) — matrix operand order repaired and checked by reading, not compiled, not run on hardware
//
#include <hip/hip_runtime.h>

// ---------------------------------------------------------------------------
// GAT (2-layer, 4-head) for MI455X / gfx1250, wave32, WMMA bf16 pipeline.
// Roofline: ~17 GFLOP dense; dominant traffic is adj (64MB). Layer-1 attention
// fuses all 4 heads per adjacency tile -> adj read 2x total (128MB ~ 5.5us at
// 23.3 TB/s). All GEMMs on v_wmma_f32_16x16x32_bf16; attention probabilities
// staged in LDS; feature-GEMM A-tiles staged via global_load_async_to_lds.
// ---------------------------------------------------------------------------

typedef __attribute__((ext_vector_type(16))) __bf16 v16bf;
typedef __attribute__((ext_vector_type(8)))  __bf16 v8bf;
typedef __attribute__((ext_vector_type(8)))  float  v8f;

#define BB    64
#define NN    512
#define COMP  128
#define GATF  64
#define HEADS 4
#define ALPHA 0.2f
#define NEGV  (-9e15f)

__device__ __forceinline__ v16bf cat16(v8bf lo, v8bf hi) {
  return __builtin_shufflevector(lo, hi, 0,1,2,3,4,5,6,7,8,9,10,11,12,13,14,15);
}

// A-fragment (16x32 bf16), row-major source, row stride in halves.
// ISA layout: h=lane>>4, elements 0..7 = K h*8..h*8+7, 8..15 = K 16+h*8..+7
__device__ __forceinline__ v16bf load_a_frag(const __bf16* base, int stride,
                                             int k0, int lane) {
  int m = lane & 15, h = lane >> 4;
  const __bf16* p = base + (size_t)m * stride + k0 + h * 8;
  v8bf lo = *(const v8bf*)(p);
  v8bf hi = *(const v8bf*)(p + 16);
  return cat16(lo, hi);
}

// B-fragment (32x16 bf16): source column-major (each output column contiguous
// in K). Same layout as A: elements 0..7 = K h*8..h*8+7, 8..15 = K 16+h*8..+7
__device__ __forceinline__ v16bf load_b_frag(const __bf16* base, int stride,
                                             int k0, int lane) {
  int n = lane & 15, h = lane >> 4;
  const __bf16* p = base + (size_t)n * stride + k0 + h * 8;
  v8bf lo = *(const v8bf*)(p);
  v8bf hi = *(const v8bf*)(p + 16);
  return cat16(lo, hi);
}

// CDNA5 async global->LDS copy (ASYNCcnt-tracked; no VGPR data round-trip).
// LDS byte offset = low 32 bits of the generic pointer (aperture rule).
__device__ __forceinline__ void async_copy16(void* lds_ptr, const void* gptr) {
  unsigned lds = (unsigned)(size_t)lds_ptr;
  asm volatile("global_load_async_to_lds_b128 %0, %1, off"
               :: "v"(lds), "v"(gptr) : "memory");
}
__device__ __forceinline__ void wait_async0() {
  asm volatile("s_wait_asynccnt 0x0" ::: "memory");
}

__device__ __forceinline__ float warp_max(float x) {
  for (int off = 16; off; off >>= 1) x = fmaxf(x, __shfl_xor(x, off, 32));
  return x;
}
__device__ __forceinline__ float warp_sum(float x) {
  for (int off = 16; off; off >>= 1) x += __shfl_xor(x, off, 32);
  return x;
}

// ---- 1. pack/transpose weights to bf16 -------------------------------------
// wT1[head*64+f][c] = W_heads[head][c][f]   (256 x 128)
// woT[fo][k]        = W_out[k][fo]          (128 x 256)
__global__ __launch_bounds__(256) void prep_kernel(
    const float* __restrict__ W_heads, const float* __restrict__ W_out,
    __bf16* __restrict__ wT1, __bf16* __restrict__ woT) {
  int t = blockIdx.x * 256 + threadIdx.x;          // 32768 threads
  if (t < 256 * 128) {
    int fg = t >> 7, c = t & 127;
    int head = fg >> 6, f = fg & 63;
    wT1[t] = (__bf16)W_heads[(head * COMP + c) * GATF + f];
  }
  if (t < 128 * 256) {
    int fo = t >> 8, k = t & 255;
    woT[t] = (__bf16)W_out[k * COMP + fo];
  }
}

// ---- 2. embedding gather -> bf16 -------------------------------------------
__global__ __launch_bounds__(128) void embed_kernel(
    const int* __restrict__ atoms, const float* __restrict__ emb,
    __bf16* __restrict__ hbf) {
  int idx = blockIdx.x;                 // b*N + n
  int tok = atoms[idx];
  hbf[(size_t)idx * COMP + threadIdx.x] =
      (__bf16)emb[(size_t)tok * COMP + threadIdx.x];
}

// ---- 3. Wh = h @ W (all heads fused, N-dim = 256) --------------------------
// out: whT[b][fg(256)][n(512)] bf16 (transposed for attention B-fragments)
__global__ __launch_bounds__(256) void gemm1_kernel(
    const __bf16* __restrict__ hbf, const __bf16* __restrict__ wT1,
    __bf16* __restrict__ whT) {
  __shared__ __bf16 shA[16][136];       // stride 68 dwords == 4 (mod 64)
  int b = blockIdx.y, n0 = blockIdx.x * 16;
  int tid = threadIdx.x, lane = tid & 31, wave = tid >> 5;
  {
    int r = tid >> 4, c8 = (tid & 15) * 8;
    async_copy16(&shA[r][c8],
                 hbf + ((size_t)b * NN + n0 + r) * COMP + c8);
    wait_async0();
  }
  __syncthreads();
  int h = lane >> 4;
  for (int q = 0; q < 2; ++q) {
    int colt = wave * 2 + q;            // 16 col tiles of 16 over 256 features
    v8f c = {};
    for (int t = 0; t < COMP / 32; ++t) {
      int k0 = t * 32;
      v16bf a  = load_a_frag(&shA[0][0], 136, k0, lane);
      v16bf bm = load_b_frag(wT1 + (size_t)colt * 16 * COMP, COMP, k0, lane);
      c = __builtin_amdgcn_wmma_f32_16x16x32_bf16(false, a, false, bm,
                                                  (short)0, c, false, false);
    }
    int fg = colt * 16 + (lane & 15);
    v8bf pk;
    for (int v = 0; v < 8; ++v) pk[v] = (__bf16)c[v];   // rows m=v+8h contig
    *(v8bf*)(whT + ((size_t)b * 256 + fg) * NN + n0 + h * 8) = pk;
  }
}

// ---- 4. f_src / f_dst (layer 1) --------------------------------------------
__global__ __launch_bounds__(512) void f1_kernel(
    const __bf16* __restrict__ whT, const float* __restrict__ a_heads,
    float* __restrict__ f1s, float* __restrict__ f1d) {
  int b = blockIdx.x >> 2, head = blockIdx.x & 3, n = threadIdx.x;
  float fs = 0.f, fd = 0.f;
  const __bf16* base = whT + ((size_t)b * 256 + head * GATF) * NN + n;
  for (int f = 0; f < GATF; ++f) {
    float w = (float)base[(size_t)f * NN];
    fs += w * a_heads[head * 2 * GATF + f];
    fd += w * a_heads[head * 2 * GATF + GATF + f];
  }
  f1s[(size_t)blockIdx.x * NN + n] = fs;
  f1d[(size_t)blockIdx.x * NN + n] = fd;
}

// ---- 5. f2 (layer 2) -------------------------------------------------------
__global__ __launch_bounds__(512) void f2_kernel(
    const __bf16* __restrict__ whoT, const float* __restrict__ a_out,
    float* __restrict__ f2s, float* __restrict__ f2d) {
  int b = blockIdx.x, n = threadIdx.x;
  float fs = 0.f, fd = 0.f;
  const __bf16* base = whoT + (size_t)b * COMP * NN + n;
  for (int f = 0; f < COMP; ++f) {
    float w = (float)base[(size_t)f * NN];
    fs += w * a_out[f];
    fd += w * a_out[COMP + f];
  }
  f2s[(size_t)b * NN + n] = fs;
  f2d[(size_t)b * NN + n] = fd;
}

// ---- 6. layer-1 attention, all heads fused per adjacency tile --------------
// Block: 128 threads (4 waves). Stages 16x512 adj tile once (bytes, 8KB LDS),
// then per head: softmax rows in registers -> bf16 att tile in LDS -> WMMA.
__global__ __launch_bounds__(128) void attn1_kernel(
    const int* __restrict__ adj, const float* __restrict__ f1s,
    const float* __restrict__ f1d, const __bf16* __restrict__ whT,
    __bf16* __restrict__ multi) {
  __shared__ unsigned char sh_adj[16][NN];       // 8 KB
  __shared__ __bf16 sh_att[16][NN + 8];          // stride 260 dw == 4 (mod 64)

  int tid = threadIdx.x, lane = tid & 31, wave = tid >> 5;
  int b = blockIdx.y, i0 = blockIdx.x * 16;

  for (int rr = 0; rr < 4; ++rr) {
    int r = wave * 4 + rr;
    const int* arow = adj + ((size_t)b * NN + i0 + r) * NN;
    for (int j = lane; j < NN; j += 32) {
      if (j + 256 < NN) __builtin_prefetch(arow + j + 256, 0, 0);
      sh_adj[r][j] = (unsigned char)(arow[j] > 0);
    }
  }
  __syncthreads();

  for (int head = 0; head < HEADS; ++head) {
    const float* fs = f1s + ((size_t)b * HEADS + head) * NN;
    const float* fd = f1d + ((size_t)b * HEADS + head) * NN;
    // softmax rows: 4 rows per wave, e-values in registers
    for (int rr = 0; rr < 4; ++rr) {
      int r = wave * 4 + rr;
      float fsi = fs[i0 + r];
      float ev[16], mx = -3.4e38f;
#pragma unroll
      for (int u = 0; u < 16; ++u) {
        int j = lane + 32 * u;
        float e = fsi + fd[j];
        e = e > 0.f ? e : ALPHA * e;
        e = sh_adj[r][j] ? e : NEGV;
        ev[u] = e;
        mx = fmaxf(mx, e);
      }
      mx = warp_max(mx);
      float sum = 0.f;
#pragma unroll
      for (int u = 0; u < 16; ++u) {
        float p = __expf(ev[u] - mx);
        ev[u] = p;
        sum += p;
      }
      sum = warp_sum(sum);
      float inv = 1.0f / sum;
#pragma unroll
      for (int u = 0; u < 16; ++u)
        sh_att[r][lane + 32 * u] = (__bf16)(ev[u] * inv);
    }
    __syncthreads();

    // [16 x 512] att @ [512 x 16] Wh tile per wave (head's 64 feats / 4 waves)
    const __bf16* whT_b = whT + ((size_t)b * 256 + head * GATF) * NN;
    int fbase = wave * 16;
    v8f c = {};
    for (int t = 0; t < NN / 32; ++t) {
      int k0 = t * 32;
      v16bf a  = load_a_frag(&sh_att[0][0], NN + 8, k0, lane);
      v16bf bm = load_b_frag(whT_b + (size_t)fbase * NN, NN, k0, lane);
      c = __builtin_amdgcn_wmma_f32_16x16x32_bf16(false, a, false, bm,
                                                  (short)0, c, false, false);
    }
    int fcol = fbase + (lane & 15), h = lane >> 4;
    for (int v = 0; v < 8; ++v) {
      int i = i0 + h * 8 + v;
      float x = c[v];
      x = x > 0.f ? x : (__expf(x) - 1.0f);    // elu (concat branch)
      multi[((size_t)b * NN + i) * 256 + head * GATF + fcol] = (__bf16)x;
    }
    __syncthreads();   // before next head overwrites sh_att
  }
}

// ---- 7. layer-2 feature GEMM: whoT = (multi @ W_out)^T ---------------------
__global__ __launch_bounds__(256) void gemm2_kernel(
    const __bf16* __restrict__ multi, const __bf16* __restrict__ woT,
    __bf16* __restrict__ whoT) {
  __shared__ __bf16 shA[16][264];       // stride 132 dwords == 4 (mod 64)
  int b = blockIdx.y, n0 = blockIdx.x * 16;
  int tid = threadIdx.x, lane = tid & 31, wave = tid >> 5;
  {
    int r = tid >> 4, c0 = (tid & 15) * 16;
    const __bf16* src = multi + ((size_t)b * NN + n0 + r) * 256 + c0;
    async_copy16(&shA[r][c0], src);
    async_copy16(&shA[r][c0 + 8], src + 8);
    wait_async0();
  }
  __syncthreads();
  int h = lane >> 4;
  v8f c = {};
  for (int t = 0; t < 256 / 32; ++t) {
    int k0 = t * 32;
    v16bf a  = load_a_frag(&shA[0][0], 264, k0, lane);
    v16bf bm = load_b_frag(woT + (size_t)wave * 16 * 256, 256, k0, lane);
    c = __builtin_amdgcn_wmma_f32_16x16x32_bf16(false, a, false, bm,
                                                (short)0, c, false, false);
  }
  int fg = wave * 16 + (lane & 15);
  v8bf pk;
  for (int v = 0; v < 8; ++v) pk[v] = (__bf16)c[v];
  *(v8bf*)(whoT + ((size_t)b * COMP + fg) * NN + n0 + h * 8) = pk;
}

// ---- 8. layer-2 attention + final elu -> f32 output ------------------------
__global__ __launch_bounds__(256) void attn2_kernel(
    const int* __restrict__ adj, const float* __restrict__ f2s,
    const float* __restrict__ f2d, const __bf16* __restrict__ whoT,
    float* __restrict__ out) {
  __shared__ __bf16 sh_att[16][NN + 8];
  int tid = threadIdx.x, lane = tid & 31, wave = tid >> 5;
  int b = blockIdx.y, i0 = blockIdx.x * 16;
  const float* fs = f2s + (size_t)b * NN;
  const float* fd = f2d + (size_t)b * NN;

  for (int rr = 0; rr < 2; ++rr) {              // 8 waves x 2 rows
    int r = wave * 2 + rr;
    const int* arow = adj + ((size_t)b * NN + i0 + r) * NN;
    float fsi = fs[i0 + r];
    float ev[16], mx = -3.4e38f;
#pragma unroll
    for (int u = 0; u < 16; ++u) {
      int j = lane + 32 * u;
      int av = arow[j];
      float e = fsi + fd[j];
      e = e > 0.f ? e : ALPHA * e;
      e = av > 0 ? e : NEGV;
      ev[u] = e;
      mx = fmaxf(mx, e);
    }
    mx = warp_max(mx);
    float sum = 0.f;
#pragma unroll
    for (int u = 0; u < 16; ++u) {
      float p = __expf(ev[u] - mx);
      ev[u] = p;
      sum += p;
    }
    sum = warp_sum(sum);
    float inv = 1.0f / sum;
#pragma unroll
    for (int u = 0; u < 16; ++u)
      sh_att[r][lane + 32 * u] = (__bf16)(ev[u] * inv);
  }
  __syncthreads();

  const __bf16* whT_b = whoT + (size_t)b * COMP * NN;
  int fbase = wave * 16;
  v8f c = {};
  for (int t = 0; t < NN / 32; ++t) {
    int k0 = t * 32;
    v16bf a  = load_a_frag(&sh_att[0][0], NN + 8, k0, lane);
    v16bf bm = load_b_frag(whT_b + (size_t)fbase * NN, NN, k0, lane);
    c = __builtin_amdgcn_wmma_f32_16x16x32_bf16(false, a, false, bm,
                                                (short)0, c, false, false);
  }
  int fcol = fbase + (lane & 15), h = lane >> 4;
  for (int v = 0; v < 8; ++v) {
    int i = i0 + h * 8 + v;
    float x = c[v];
    x = x > 0.f ? x : (__expf(x) - 1.0f);       // final elu
    out[((size_t)b * NN + i) * COMP + fcol] = x;
  }
}

// ---------------------------------------------------------------------------
extern "C" void kernel_launch(void* const* d_in, const int* in_sizes, int n_in,
                              void* d_out, int out_size, void* d_ws,
                              size_t ws_size, hipStream_t stream) {
  (void)in_sizes; (void)n_in; (void)out_size; (void)ws_size;
  const int*   atoms   = (const int*)d_in[0];
  const int*   adj     = (const int*)d_in[1];
  const float* emb     = (const float*)d_in[2];
  const float* W_heads = (const float*)d_in[3];
  const float* a_heads = (const float*)d_in[4];
  const float* W_out   = (const float*)d_in[5];
  const float* a_out   = (const float*)d_in[6];
  float* out = (float*)d_out;

  char* w = (char*)d_ws;
  __bf16* hbf  = (__bf16*)w; w += (size_t)BB * NN * COMP * 2;     // 8 MB
  __bf16* wT1  = (__bf16*)w; w += (size_t)256 * COMP * 2;
  __bf16* woT  = (__bf16*)w; w += (size_t)COMP * 256 * 2;
  __bf16* whT  = (__bf16*)w; w += (size_t)BB * 256 * NN * 2;      // 16 MB
  float*  f1s  = (float*)w;  w += (size_t)BB * HEADS * NN * 4;
  float*  f1d  = (float*)w;  w += (size_t)BB * HEADS * NN * 4;
  __bf16* mult = (__bf16*)w; w += (size_t)BB * NN * 256 * 2;      // 16 MB
  __bf16* whoT = (__bf16*)w; w += (size_t)BB * COMP * NN * 2;     // 8 MB
  float*  f2s  = (float*)w;  w += (size_t)BB * NN * 4;
  float*  f2d  = (float*)w;  w += (size_t)BB * NN * 4;

  prep_kernel<<<128, 256, 0, stream>>>(W_heads, W_out, wT1, woT);
  embed_kernel<<<BB * NN, 128, 0, stream>>>(atoms, emb, hbf);
  gemm1_kernel<<<dim3(NN / 16, BB), 256, 0, stream>>>(hbf, wT1, whT);
  f1_kernel<<<BB * HEADS, 512, 0, stream>>>(whT, a_heads, f1s, f1d);
  attn1_kernel<<<dim3(NN / 16, BB), 128, 0, stream>>>(adj, f1s, f1d, whT,
                                                      mult);
  gemm2_kernel<<<dim3(NN / 16, BB), 256, 0, stream>>>(mult, woT, whoT);
  f2_kernel<<<BB, 512, 0, stream>>>(whoT, a_out, f2s, f2d);
  attn2_kernel<<<dim3(NN / 16, BB), 256, 0, stream>>>(adj, f2s, f2d, whoT,
                                                      out);
}
